// TransformerProgressHead_69965017252585
// MI455X (gfx1250) — matrix operand order repaired and checked by reading, not compiled, not run on hardware
//
#include <hip/hip_runtime.h>

// ---------------------------------------------------------------------------
// Types
// ---------------------------------------------------------------------------
typedef __bf16  v16bf __attribute__((ext_vector_type(16)));
typedef float   v8f   __attribute__((ext_vector_type(8)));
typedef float   f4v   __attribute__((ext_vector_type(4)));
typedef unsigned int  u32a  __attribute__((may_alias));
typedef unsigned int  uint2v __attribute__((ext_vector_type(2)));
typedef unsigned int  uint4v __attribute__((ext_vector_type(4)));
typedef uint2v  u64a  __attribute__((may_alias));
typedef uint4v  u128a __attribute__((may_alias));
typedef f4v     f4a   __attribute__((may_alias));

union Frag { u32a u[8]; v16bf v; };

// Problem constants (match reference)
#define T_LEN 2048
#define IN_D  1024
#define D_MOD 1024
#define N_H   16
#define HD    64
#define N_L   4
#define F_FFN 4096

// ---------------------------------------------------------------------------
// Helpers
// ---------------------------------------------------------------------------
__device__ __forceinline__ unsigned short f2bf(float f) {
  unsigned int u = __float_as_uint(f);
  unsigned int r = (u + 0x7FFFu + ((u >> 16) & 1u)) >> 16;
  return (unsigned short)r;
}

__device__ __forceinline__ v8f wmma_bf16(v16bf a, v16bf b, v8f c) {
  // D(16x16 f32) = A(16x32 bf16) * B(32x16 bf16) + C
  return __builtin_amdgcn_wmma_f32_16x16x32_bf16(
      false, a, false, b, (short)0, c, false, false);
}

// A-matrix 16x32 bf16 fragment from row-major LDS tile [rows][ld]
// lane l: row = row0 + (l&15); k-base = (l>>4)*8; VGPR j<4: K=kb+2j,+1;
// VGPR j>=4: K=kb+16+2(j-4),+1   (per CDNA5 ISA 7.12.2)
__device__ __forceinline__ v16bf lds_frag_a(const unsigned short* p0, int ld,
                                            int row0, int k0, int lane) {
  const unsigned short* p = p0 + (row0 + (lane & 15)) * ld + k0 + ((lane >> 4) << 3);
  Frag f;
#pragma unroll
  for (int j = 0; j < 4; ++j) {
    f.u[j]     = *(const u32a*)(p + 2 * j);
    f.u[j + 4] = *(const u32a*)(p + 16 + 2 * j);
  }
  return f.v;
}

// B-matrix 32x16 bf16 fragment from TRANSPOSED LDS storage:
// element (k, n) stored at p0[n*ld + k] -> contiguous dwords, ds_load_b128 pairs
// lane l: col n = n0 + (l&15); k-base = (l>>4)*8; VGPR j<4: K=kb+2j,+1; VGPR j>=4: K=kb+16+2(j-4),+1
__device__ __forceinline__ v16bf lds_frag_bt(const unsigned short* p0, int ld,
                                             int k0, int n0, int lane) {
  const unsigned short* p = p0 + (n0 + (lane & 15)) * ld + k0 + ((lane >> 4) << 3);
  Frag f;
#pragma unroll
  for (int j = 0; j < 4; ++j) {
    f.u[j]     = *(const u32a*)(p + 2 * j);
    f.u[j + 4] = *(const u32a*)(p + 16 + 2 * j);
  }
  return f.v;
}

// ---------------------------------------------------------------------------
// fp32 -> bf16 cast (activations), vectorized x4, grid-stride
// ---------------------------------------------------------------------------
__global__ __launch_bounds__(256) void cast_f2bf_kernel(const float* __restrict__ in,
                                                        unsigned short* __restrict__ out,
                                                        long long n4) {  // n/4
  long long i = (long long)blockIdx.x * 256 + threadIdx.x;
  long long stride = (long long)gridDim.x * 256;
  for (; i < n4; i += stride) {
    f4v v = *(const f4a*)(in + i * 4);
    uint2v p;
    p.x = (unsigned int)f2bf(v.x) | ((unsigned int)f2bf(v.y) << 16);
    p.y = (unsigned int)f2bf(v.z) | ((unsigned int)f2bf(v.w) << 16);
    *(u64a*)(out + i * 4) = p;
  }
}

// ---------------------------------------------------------------------------
// Transposing cast for weights: in fp32 [l][K][N] -> out bf16 [l][N][K]
// 32x32 tiles through LDS; coalesced on both sides.
// ---------------------------------------------------------------------------
__global__ __launch_bounds__(256) void castT_kernel(const float* __restrict__ in,
                                                    unsigned short* __restrict__ out,
                                                    int K, int N) {
  __shared__ float tile[32][33];
  const int bk = blockIdx.x * 32, bn = blockIdx.y * 32;
  const size_t slab = (size_t)blockIdx.z * K * N;
  in  += slab;
  out += slab;
  const int tx = threadIdx.x & 31, ty = threadIdx.x >> 5;  // 8 rows per pass
#pragma unroll
  for (int r = ty; r < 32; r += 8)
    tile[r][tx] = in[(size_t)(bk + r) * N + bn + tx];
  __syncthreads();
#pragma unroll
  for (int r = ty; r < 32; r += 8)
    out[(size_t)(bn + r) * K + bk + tx] = f2bf(tile[tx][r]);
}

// ---------------------------------------------------------------------------
// Tiled bf16 GEMM:  C[M,N] = A[M,K] @ B[K,N] + bias (+ res) (+ GELU)
// B supplied TRANSPOSED (Bt = [N][K] bf16). Block tile 128x128, BK=32.
// 8 waves (2 x 4); wave owns 64x32 = 8 WMMA tiles per k-step.
// Double-buffered LDS (one barrier per k-step); register prefetch of the
// next tile overlaps the WMMAs; global_prefetch issued 2 k-steps ahead.
// OUT_BT: write bf16 output transposed ([N][M], packed b128 stores) — used
// to produce V^T directly for attention.
// ---------------------------------------------------------------------------
template <bool HAS_BIAS, bool HAS_RES, bool DO_GELU, bool OUT_F, bool OUT_B, bool OUT_BT>
__global__ __launch_bounds__(256) void gemm_bf16_kernel(
    const unsigned short* __restrict__ A, const unsigned short* __restrict__ Bt,
    const float* __restrict__ bias, const float* __restrict__ res,
    float* __restrict__ outF, unsigned short* __restrict__ outB,
    int M, int N, int K) {
  __shared__ unsigned short As[2][128 * 32];   // [m][k]
  __shared__ unsigned short Bs[2][128 * 32];   // [n][k]  (transposed storage)

  const int tid  = threadIdx.x;
  const int lane = tid & 31;
  const int w    = tid >> 5;
  const int wr   = w & 1;    // wave row (0..1) -> 64 rows each
  const int wc   = w >> 1;   // wave col (0..3) -> 32 cols each
  const int m0   = blockIdx.y * 128;
  const int n0   = blockIdx.x * 128;

  // Staging addresses (constant over k-loop)
  const int eA = tid * 16, rA = eA >> 5, cA = eA & 31;
  const unsigned short* gA = A  + (size_t)(m0 + rA) * K + cA;
  const unsigned short* gB = Bt + (size_t)(n0 + rA) * K + cA;  // same tiling

  v8f acc[4][2] = {};
  uint4v pa0, pa1, pb0, pb1;

  // Prologue: stage k-tile 0 into buffer 0
  pa0 = *(const u128a*)(gA);
  pa1 = *(const u128a*)(gA + 8);
  pb0 = *(const u128a*)(gB);
  pb1 = *(const u128a*)(gB + 8);
  *(u128a*)(As[0] + eA) = pa0;  *(u128a*)(As[0] + eA + 8) = pa1;
  *(u128a*)(Bs[0] + eA) = pb0;  *(u128a*)(Bs[0] + eA + 8) = pb1;
  __syncthreads();

  int cur = 0;
  for (int kt = 0; kt < K; kt += 32) {
    const bool more = (kt + 32) < K;

    v16bf aF[4], bF[2];
#pragma unroll
    for (int mi = 0; mi < 4; ++mi)
      aF[mi] = lds_frag_a(As[cur], 32, wr * 64 + mi * 16, 0, lane);
#pragma unroll
    for (int ni = 0; ni < 2; ++ni)
      bF[ni] = lds_frag_bt(Bs[cur], 32, 0, wc * 32 + ni * 16, lane);

    if (more) {  // register prefetch of next tile; overlaps the WMMAs below
      pa0 = *(const u128a*)(gA + kt + 32);
      pa1 = *(const u128a*)(gA + kt + 40);
      pb0 = *(const u128a*)(gB + kt + 32);
      pb1 = *(const u128a*)(gB + kt + 40);
      if (kt + 64 < K) {  // L2 prefetch 2 k-steps ahead (global_prefetch_b8)
        __builtin_prefetch(gA + kt + 64, 0, 1);
        __builtin_prefetch(gB + kt + 64, 0, 1);
      }
    }

#pragma unroll
    for (int mi = 0; mi < 4; ++mi)
#pragma unroll
      for (int ni = 0; ni < 2; ++ni)
        acc[mi][ni] = wmma_bf16(aF[mi], bF[ni], acc[mi][ni]);

    if (more) {
      const int nxt = cur ^ 1;
      *(u128a*)(As[nxt] + eA) = pa0;  *(u128a*)(As[nxt] + eA + 8) = pa1;
      *(u128a*)(Bs[nxt] + eA) = pb0;  *(u128a*)(Bs[nxt] + eA + 8) = pb1;
      __syncthreads();
      cur = nxt;
    }
  }

  // Epilogue. C layout: lane l -> col (l&15); VGPR j -> row j + 8*(l>>4)
  if (OUT_BT) {
    // Transposed bf16 output [N][M]: the 8 rows per lane are consecutive,
    // so each tile packs into one 16-byte store per lane.
#pragma unroll
    for (int mi = 0; mi < 4; ++mi)
#pragma unroll
      for (int ni = 0; ni < 2; ++ni) {
        const int col  = n0 + wc * 32 + ni * 16 + (lane & 15);
        const int row0 = m0 + wr * 64 + mi * 16 + ((lane >> 4) << 3);
        const float b = HAS_BIAS ? bias[col] : 0.f;
        uint4v pk;
        pk.x = (unsigned int)f2bf(acc[mi][ni][0] + b) |
               ((unsigned int)f2bf(acc[mi][ni][1] + b) << 16);
        pk.y = (unsigned int)f2bf(acc[mi][ni][2] + b) |
               ((unsigned int)f2bf(acc[mi][ni][3] + b) << 16);
        pk.z = (unsigned int)f2bf(acc[mi][ni][4] + b) |
               ((unsigned int)f2bf(acc[mi][ni][5] + b) << 16);
        pk.w = (unsigned int)f2bf(acc[mi][ni][6] + b) |
               ((unsigned int)f2bf(acc[mi][ni][7] + b) << 16);
        *(u128a*)(outB + (size_t)col * M + row0) = pk;
      }
    return;
  }

#pragma unroll
  for (int mi = 0; mi < 4; ++mi)
#pragma unroll
    for (int j = 0; j < 8; ++j) {
      const int row = m0 + wr * 64 + mi * 16 + j + ((lane >> 4) << 3);
      const size_t rowBase = (size_t)row * N;
#pragma unroll
      for (int ni = 0; ni < 2; ++ni) {
        const int col = n0 + wc * 32 + ni * 16 + (lane & 15);
        float v = acc[mi][ni][j];
        if (HAS_BIAS) v += bias[col];
        const size_t idx = rowBase + col;
        if (HAS_RES)  v += res[idx];
        if (DO_GELU)  v = 0.5f * v * (1.0f + erff(v * 0.70710678118654752f));
        if (OUT_F) outF[idx] = v;
        if (OUT_B) outB[idx] = f2bf(v);
      }
    }
}

// ---------------------------------------------------------------------------
// Flash attention (causal + ALiBi), bf16 Q/K/V with fp32 accumulation.
// Block = (head, 64-query tile). 128 threads = 4 waves; wave owns 16 q-rows.
// V is supplied transposed (Vt = [D][T]) so its fragments are contiguous.
// Fragment loads are batched per k-step so multiple ds_load_b128 stay in
// flight ahead of the WMMA group.
// ---------------------------------------------------------------------------
__global__ __launch_bounds__(128) void attn_kernel(
    const unsigned short* __restrict__ Q, const unsigned short* __restrict__ Kb,
    const unsigned short* __restrict__ Vt, unsigned short* __restrict__ O,
    int T, int D) {
  __shared__ unsigned short Qs[64 * 64];   // [q][hd]
  __shared__ unsigned short Ks[64 * 64];   // [key][hd]  (= B^T storage for QK^T)
  __shared__ unsigned short Vs[64 * 64];   // [hd][key]  (= B^T storage for P*V)
  __shared__ unsigned short Ps[4][16 * 64];

  const int tid  = threadIdx.x;
  const int lane = tid & 31;
  const int wq   = tid >> 5;          // wave = query sub-tile 0..3
  const int qb0  = blockIdx.x * 64;
  const int head = blockIdx.y;
  const int col0 = head * HD;
  const float slope    = exp2f(-(float)head);   // 1/2^h
  const float invScale = 0.125f;                // 1/sqrt(64)

  // Load Q tile (64 rows x 64 cols) into LDS
  for (int e = tid * 8; e < 64 * 64; e += 128 * 8) {
    int r = e >> 6, c = e & 63;
    *(u128a*)(Qs + e) = *(const u128a*)(Q + (size_t)(qb0 + r) * D + col0 + c);
  }
  __syncthreads();

  v16bf qA[2];
  qA[0] = lds_frag_a(Qs, 64, wq * 16, 0,  lane);
  qA[1] = lds_frag_a(Qs, 64, wq * 16, 32, lane);

  v8f accO[4] = {};
  float rowMax[8], rowSum[8];
#pragma unroll
  for (int j = 0; j < 8; ++j) { rowMax[j] = -3.0e38f; rowSum[j] = 0.f; }

  for (int kb = 0; kb <= qb0; kb += 64) {
    __syncthreads();
    for (int e = tid * 8; e < 64 * 64; e += 128 * 8) {
      int r = e >> 6, c = e & 63;
      *(u128a*)(Ks + e) = *(const u128a*)(Kb + (size_t)(kb + r) * D + col0 + c);
      // Vt rows are head-dim channels; columns are time -> already B^T order
      *(u128a*)(Vs + e) = *(const u128a*)(Vt + (size_t)(col0 + r) * T + kb + c);
      if (kb + 64 <= qb0) {  // prefetch next key-block (global_prefetch_b8)
        __builtin_prefetch(Kb + (size_t)(kb + 64 + r) * D + col0 + c, 0, 1);
        __builtin_prefetch(Vt + (size_t)(col0 + r) * T + kb + 64 + c, 0, 1);
      }
    }
    __syncthreads();

    // S = Q @ K^T : 16 rows x 64 keys, contract HD=64 (2 k-steps).
    // Batch the 4 B-fragment loads ahead of the 4 WMMAs.
    v8f s[4] = {};
#pragma unroll
    for (int kk = 0; kk < 2; ++kk) {
      v16bf bt[4];
#pragma unroll
      for (int nt = 0; nt < 4; ++nt)
        bt[nt] = lds_frag_bt(Ks, 64, kk * 32, nt * 16, lane);
#pragma unroll
      for (int nt = 0; nt < 4; ++nt)
        s[nt] = wmma_bf16(qA[kk], bt[nt], s[nt]);
    }

    // Streaming softmax in C layout (half-wave owns each row)
#pragma unroll
    for (int j = 0; j < 8; ++j) {
      const int irow = qb0 + wq * 16 + j + ((lane >> 4) << 3);
      float sv[4];
      float tmax = -3.0e38f;
#pragma unroll
      for (int nt = 0; nt < 4; ++nt) {
        int jcol = kb + nt * 16 + (lane & 15);
        float v = s[nt][j] * invScale - slope * (float)(irow - jcol);
        if (jcol > irow) v = -1.0e30f;
        sv[nt] = v;
        tmax   = fmaxf(tmax, v);
      }
#pragma unroll
      for (int o = 1; o < 16; o <<= 1)
        tmax = fmaxf(tmax, __shfl_xor(tmax, o, 32));
      float mnew = fmaxf(rowMax[j], tmax);
      float corr = expf(rowMax[j] - mnew);
      float psum = 0.f;
#pragma unroll
      for (int nt = 0; nt < 4; ++nt) {
        float p = expf(sv[nt] - mnew);
        psum += p;
        Ps[wq][(j + ((lane >> 4) << 3)) * 64 + nt * 16 + (lane & 15)] = f2bf(p);
      }
#pragma unroll
      for (int o = 1; o < 16; o <<= 1) psum += __shfl_xor(psum, o, 32);
      rowSum[j] = rowSum[j] * corr + psum;
      rowMax[j] = mnew;
#pragma unroll
      for (int nt = 0; nt < 4; ++nt) accO[nt][j] *= corr;
    }

    // O += P @ V : contract over 64 keys (2 k-steps); Vs is B^T storage.
#pragma unroll
    for (int kk = 0; kk < 2; ++kk) {
      v16bf aP = lds_frag_a(Ps[wq], 64, 0, kk * 32, lane);
      v16bf bV[4];
#pragma unroll
      for (int nt = 0; nt < 4; ++nt)
        bV[nt] = lds_frag_bt(Vs, 64, kk * 32, nt * 16, lane);
#pragma unroll
      for (int nt = 0; nt < 4; ++nt)
        accO[nt] = wmma_bf16(aP, bV[nt], accO[nt]);
    }
  }

  // Normalize and write bf16 output [T, D]
#pragma unroll
  for (int nt = 0; nt < 4; ++nt)
#pragma unroll
    for (int j = 0; j < 8; ++j) {
      int row = qb0 + wq * 16 + j + ((lane >> 4) << 3);
      int col = col0 + nt * 16 + (lane & 15);
      O[(size_t)row * D + col] = f2bf(accO[nt][j] / rowSum[j]);
    }
}

// ---------------------------------------------------------------------------
// LayerNorm per row (fp32 in; optional fp32 and/or bf16 out)
// ---------------------------------------------------------------------------
__global__ __launch_bounds__(256) void ln_kernel(const float* __restrict__ in,
                                                 const float* __restrict__ g,
                                                 const float* __restrict__ b,
                                                 float* __restrict__ outF,
                                                 unsigned short* __restrict__ outB,
                                                 int D) {
  __shared__ float rs[256], rq[256];
  const int row = blockIdx.x, t = threadIdx.x;
  const float* p = in + (size_t)row * D;
  float s = 0.f, q = 0.f;
  for (int i = t; i < D; i += 256) { float v = p[i]; s += v; q += v * v; }
  rs[t] = s; rq[t] = q;
  __syncthreads();
  for (int o = 128; o > 0; o >>= 1) {
    if (t < o) { rs[t] += rs[t + o]; rq[t] += rq[t + o]; }
    __syncthreads();
  }
  float mean = rs[0] / (float)D;
  float var  = rq[0] / (float)D - mean * mean;
  float inv  = rsqrtf(var + 1e-5f);
  for (int i = t; i < D; i += 256) {
    float v = (p[i] - mean) * inv * g[i] + b[i];
    size_t idx = (size_t)row * D + i;
    if (outF) outF[idx] = v;
    if (outB) outB[idx] = f2bf(v);
  }
}

// ---------------------------------------------------------------------------
// Final head: relu(x[T-1] @ o1W + o1b) @ o2W + o2b -> sigmoid
// ---------------------------------------------------------------------------
__global__ __launch_bounds__(256) void head_kernel(const float* __restrict__ x,
                                                   const float* __restrict__ o1W,
                                                   const float* __restrict__ o1b,
                                                   const float* __restrict__ o2W,
                                                   const float* __restrict__ o2b,
                                                   float* __restrict__ out,
                                                   int T, int D) {
  __shared__ float part[256];
  __shared__ float h[32];
  const float* xr = x + (size_t)(T - 1) * D;
  const int t = threadIdx.x;
  const int o = t & 31, p = t >> 5;
  float s = 0.f;
  for (int d = p; d < D; d += 8) s += xr[d] * o1W[(size_t)d * 32 + o];
  part[t] = s;
  __syncthreads();
  if (t < 32) {
    float acc = o1b[t];
    for (int pp = 0; pp < 8; ++pp) acc += part[pp * 32 + t];
    h[t] = fmaxf(acc, 0.f);
  }
  __syncthreads();
  if (t == 0) {
    float acc = o2b[0];
    for (int oo = 0; oo < 32; ++oo) acc += h[oo] * o2W[oo];
    out[0] = 1.f / (1.f + expf(-acc));
  }
}

// ---------------------------------------------------------------------------
// Orchestration
// ---------------------------------------------------------------------------
extern "C" void kernel_launch(void* const* d_in, const int* in_sizes, int n_in,
                              void* d_out, int out_size, void* d_ws, size_t ws_size,
                              hipStream_t stream) {
  (void)in_sizes; (void)n_in; (void)out_size; (void)ws_size;

  const float* seg  = (const float*)d_in[0];
  const float* in_W = (const float*)d_in[1];
  const float* in_b = (const float*)d_in[2];
  const float* in_g = (const float*)d_in[3];
  const float* in_beta = (const float*)d_in[4];
  const float* qW = (const float*)d_in[5];  const float* qb = (const float*)d_in[6];
  const float* kW = (const float*)d_in[7];  const float* kb = (const float*)d_in[8];
  const float* vW = (const float*)d_in[9];  const float* vb = (const float*)d_in[10];
  const float* oW = (const float*)d_in[11]; const float* ob = (const float*)d_in[12];
  const float* n1g = (const float*)d_in[13]; const float* n1b = (const float*)d_in[14];
  const float* n2g = (const float*)d_in[15]; const float* n2b = (const float*)d_in[16];
  const float* f1W = (const float*)d_in[17]; const float* f1b = (const float*)d_in[18];
  const float* f2W = (const float*)d_in[19]; const float* f2b = (const float*)d_in[20];
  const float* o1W = (const float*)d_in[21]; const float* o1b = (const float*)d_in[22];
  const float* o2W = (const float*)d_in[23]; const float* o2b = (const float*)d_in[24];

  // Workspace carve-out
  char* base = (char*)d_ws;
  size_t off = 0;
  auto alloc = [&](size_t bytes) {
    void* p = base + off;
    off = (off + bytes + 255) & ~(size_t)255;
    return p;
  };
  typedef unsigned short us;
  us* segB  = (us*)alloc((size_t)T_LEN * IN_D * 2);
  us* inWT  = (us*)alloc((size_t)IN_D * D_MOD * 2);          // [D][IN]
  us* qWT   = (us*)alloc((size_t)N_L * D_MOD * D_MOD * 2);   // [l][D][D] (transposed)
  us* kWT   = (us*)alloc((size_t)N_L * D_MOD * D_MOD * 2);
  us* vWT   = (us*)alloc((size_t)N_L * D_MOD * D_MOD * 2);
  us* oWT   = (us*)alloc((size_t)N_L * D_MOD * D_MOD * 2);
  us* f1WT  = (us*)alloc((size_t)N_L * D_MOD * F_FFN * 2);   // [l][F][D]
  us* f2WT  = (us*)alloc((size_t)N_L * F_FFN * D_MOD * 2);   // [l][D][F]
  float* x  = (float*)alloc((size_t)T_LEN * D_MOD * 4);
  float* pre = (float*)alloc((size_t)T_LEN * D_MOD * 4);
  us* xn    = (us*)alloc((size_t)T_LEN * D_MOD * 2);
  us* qBuf  = (us*)alloc((size_t)T_LEN * D_MOD * 2);
  us* kBuf  = (us*)alloc((size_t)T_LEN * D_MOD * 2);
  us* vT    = (us*)alloc((size_t)T_LEN * D_MOD * 2);         // [D][T]
  us* attnB = (us*)alloc((size_t)T_LEN * D_MOD * 2);
  us* hB    = (us*)alloc((size_t)T_LEN * F_FFN * 2);

  // Activation cast (row-major kept)
  cast_f2bf_kernel<<<2048, 256, 0, stream>>>(seg, segB,
                                             ((long long)T_LEN * IN_D) / 4);
  // Weight casts with transpose: [K][N] fp32 -> [N][K] bf16
  auto castT = [&](const float* src, us* dst, int K, int N, int L) {
    castT_kernel<<<dim3(K / 32, N / 32, L), 256, 0, stream>>>(src, dst, K, N);
  };
  castT(in_W, inWT, IN_D, D_MOD, 1);
  castT(qW, qWT, D_MOD, D_MOD, N_L);
  castT(kW, kWT, D_MOD, D_MOD, N_L);
  castT(vW, vWT, D_MOD, D_MOD, N_L);
  castT(oW, oWT, D_MOD, D_MOD, N_L);
  castT(f1W, f1WT, D_MOD, F_FFN, N_L);
  castT(f2W, f2WT, F_FFN, D_MOD, N_L);

  // GEMM dispatch (B transposed). Epilogue variants resolved at compile time.
  // mode 0: bias -> fp32 out            (input projection)
  // mode 1: bias -> bf16 out            (Q/K projections)
  // mode 2: bias + residual -> fp32     (O-proj, FFN2)
  // mode 3: bias + GELU -> bf16         (FFN1)
  // mode 4: bias -> bf16 TRANSPOSED out (V projection -> V^T)
  auto gemm = [&](const us* A, const us* Bw, const float* bias, const float* res,
                  float* outF, us* outB, int M, int N, int K, int mode) {
    dim3 grid(N / 128, M / 128), blk(256);
    switch (mode) {
      case 0: gemm_bf16_kernel<true, false, false, true,  false, false>
                  <<<grid, blk, 0, stream>>>(A, Bw, bias, res, outF, outB, M, N, K); break;
      case 1: gemm_bf16_kernel<true, false, false, false, true,  false>
                  <<<grid, blk, 0, stream>>>(A, Bw, bias, res, outF, outB, M, N, K); break;
      case 2: gemm_bf16_kernel<true, true,  false, true,  false, false>
                  <<<grid, blk, 0, stream>>>(A, Bw, bias, res, outF, outB, M, N, K); break;
      case 3: gemm_bf16_kernel<true, false, true,  false, true,  false>
                  <<<grid, blk, 0, stream>>>(A, Bw, bias, res, outF, outB, M, N, K); break;
      default: gemm_bf16_kernel<true, false, false, false, false, true>
                  <<<grid, blk, 0, stream>>>(A, Bw, bias, res, outF, outB, M, N, K); break;
    }
  };

  // x = LN(seg @ in_W + in_b)
  gemm(segB, inWT, in_b, nullptr, pre, nullptr, T_LEN, D_MOD, IN_D, 0);
  ln_kernel<<<T_LEN, 256, 0, stream>>>(pre, in_g, in_beta, x, nullptr, D_MOD);

  for (int l = 0; l < N_L; ++l) {
    const size_t wDD = (size_t)l * D_MOD * D_MOD;
    const size_t wDF = (size_t)l * D_MOD * F_FFN;
    // xn = LN1(x)  (bf16 for WMMA)
    ln_kernel<<<T_LEN, 256, 0, stream>>>(x, n1g + l * D_MOD, n1b + l * D_MOD,
                                         nullptr, xn, D_MOD);
    // Q,K projections (bf16 out); V projection written transposed
    gemm(xn, qWT + wDD, qb + l * D_MOD, nullptr, nullptr, qBuf, T_LEN, D_MOD, D_MOD, 1);
    gemm(xn, kWT + wDD, kb + l * D_MOD, nullptr, nullptr, kBuf, T_LEN, D_MOD, D_MOD, 1);
    gemm(xn, vWT + wDD, vb + l * D_MOD, nullptr, nullptr, vT,   T_LEN, D_MOD, D_MOD, 4);
    // Flash attention with ALiBi + causal mask
    attn_kernel<<<dim3(T_LEN / 64, N_H), 128, 0, stream>>>(qBuf, kBuf, vT,
                                                           attnB, T_LEN, D_MOD);
    // x = x + attn @ oW + ob
    gemm(attnB, oWT + wDD, ob + l * D_MOD, x, x, nullptr, T_LEN, D_MOD, D_MOD, 2);
    // xn = LN2(x)
    ln_kernel<<<T_LEN, 256, 0, stream>>>(x, n2g + l * D_MOD, n2b + l * D_MOD,
                                         nullptr, xn, D_MOD);
    // h = gelu(xn @ f1W + f1b)   (bf16)
    gemm(xn, f1WT + wDF, f1b + l * F_FFN, nullptr, nullptr, hB, T_LEN, F_FFN, D_MOD, 3);
    // x = x + h @ f2W + f2b
    gemm(hB, f2WT + wDF, f2b + l * D_MOD, x, x, nullptr, T_LEN, D_MOD, F_FFN, 2);
  }

  head_kernel<<<1, 256, 0, stream>>>(x, o1W, o1b, o2W, o2b, (float*)d_out,
                                     T_LEN, D_MOD);
}
